// LoRAAWQW4A16Linear_58720792871078
// MI455X (gfx1250) — matrix operand order repaired and checked by reading, not compiled, not run on hardware
//
#include <hip/hip_runtime.h>

typedef __attribute__((ext_vector_type(16))) _Float16 v16h;
typedef __attribute__((ext_vector_type(8)))  _Float16 v8h;
typedef __attribute__((ext_vector_type(4)))  _Float16 v4h;
typedef __attribute__((ext_vector_type(8)))  float    v8f;
typedef __attribute__((ext_vector_type(4)))  float    v4f;
typedef __attribute__((ext_vector_type(4)))  int      v4i;

// Problem constants (match reference)
constexpr int kT   = 64;      // tokens
constexpr int kIN  = 4096;    // in_features
constexpr int kOUT = 12288;   // out_features
constexpr int kG   = 32;      // quant groups along IN (GS = 128)
constexpr int kR   = 16;      // LoRA rank
constexpr int kLR  = 64;      // L*R fused lora columns
constexpr int kKC  = 128;     // K-chunk == one AWQ group
constexpr int kONT = 128;     // out-channel tile per block
constexpr int kKSPLIT = 4;    // K-split slices (occupancy: 96 -> 384 blocks)

// ---------------------------------------------------------------------------
// Kernel 1: t[T, L*R] = (x @ down^T) * lora_scale[l]   (tiny: 33 MFLOP)
// ---------------------------------------------------------------------------
__global__ __launch_bounds__(256)
void lora_down_kernel(const float* __restrict__ x,
                      const float* __restrict__ down,
                      const float* __restrict__ lscale,
                      float* __restrict__ tws)
{
    __shared__ float red[256];
    const int lr  = blockIdx.x;      // 0..63  (l = lr/16, r = lr%16)
    const int tid = threadIdx.x;
    const int tt  = tid & 63;        // token
    const int ks  = tid >> 6;        // K split 0..3

    const float* xi = x    + tt * kIN + ks * (kIN / 4);
    const float* di = down + lr * kIN + ks * (kIN / 4);

    float acc = 0.f;
    #pragma unroll 4
    for (int i = 0; i < kIN / 4; i += 4) {
        v4f a = *(const v4f*)(xi + i);
        v4f b = *(const v4f*)(di + i);
        acc += a.x * b.x + a.y * b.y + a.z * b.z + a.w * b.w;
    }
    red[tid] = acc;
    __syncthreads();
    if (ks == 0) {
        float s = red[tt] + red[tt + 64] + red[tt + 128] + red[tt + 192];
        tws[tt * kLR + lr] = s * lscale[lr >> 4];
    }
}

// ---------------------------------------------------------------------------
// Kernel 2: main streaming GEMM. grid = (OUT/128, KSPLIT), 256 thr (8 waves).
// Block (bx, kz): all 64 token rows x 128 out-channels, K-chunks
// [kz*gCount, kz*gCount+gCount). Writes f32 partial tile (slice kz) of out.
// LoRA-up (K = L*R = 64) folded into slice kz==0 only.
// ---------------------------------------------------------------------------
__global__ __launch_bounds__(256)
void awq_lora_gemm(const float* __restrict__ x,
                   const int*   __restrict__ qw,
                   const int*   __restrict__ zeros,
                   const float* __restrict__ scales,
                   const float* __restrict__ up,
                   const float* __restrict__ tws,
                   float* __restrict__ out,   // [gridDim.y][kT][kOUT]
                   int gCount)
{
    __shared__ _Float16 Ash[kT * kKC];     // 64x128 f16 x-tile  (16 KB)
    __shared__ _Float16 Wsh[kONT * kKC];   // 128x128 f16 w-tile (32 KB)

    const int tid   = threadIdx.x;
    const int lane  = tid & 31;
    const int wave  = tid >> 5;
    const int obase = blockIdx.x * kONT;
    const int kz    = blockIdx.y;

    // Wave tiling: 8 waves cover 4 M-stripes x 2 N-halves; each wave does
    // 4 N-subtiles of 16 -> 32 subtiles of 16x16 total.
    const int mBlk  = (wave >> 1) * 16;    // 0,16,32,48
    const int nHalf = (wave & 1) * 64;     // 0,64

    // Per-lane fragment addressing (ISA 7.12.2, wave32):
    const int lhi  = lane >> 4;            // 0: lanes 0-15, 1: lanes 16-31
    const int ln   = lane & 15;
    const int aRow = mBlk + ln;            // A row held by this lane
    const int aKlo = lhi * 16;             // A: 16 contiguous K per lane-half
    const int bK   = lhi * 16;             // B: 16 contiguous K per lane-half

    v8f c0 = {}, c1 = {}, c2 = {}, c3 = {};

    const int g0 = kz * gCount;
    for (int gi = 0; gi < gCount; ++gi) {
        const int g   = g0 + gi;
        const int k0g = g * kKC;

        // ---- stage x tile (f32 -> f16): 32 floats per thread ----
        {
            const int row = tid >> 2;
            const int col = (tid & 3) * 32;
            const float* src = x + row * kIN + k0g + col;
            _Float16*    dst = &Ash[row * kKC + col];
            #pragma unroll
            for (int i = 0; i < 8; ++i) {
                v4f f = *(const v4f*)(src + i * 4);
                *(v4h*)(dst + i * 4) = __builtin_convertvector(f, v4h);
            }
        }
        // ---- dequant w tile: (q - zero) * scale, 64 ints per thread ----
        {
            const int wrow  = tid >> 1;
            const int wcol0 = (tid & 1) * 64;
            const int o     = obase + wrow;
            const float s   = scales[o * kG + g];
            const float z   = (float)zeros[o * kG + g];
            const int* qsrc = qw + (size_t)o * kIN + k0g + wcol0;
            _Float16*  wdst = &Wsh[wrow * kKC + wcol0];
            #pragma unroll
            for (int i = 0; i < 16; ++i) {
                v4i q = *(const v4i*)(qsrc + i * 4);
                v4f f = { ((float)q.x - z) * s, ((float)q.y - z) * s,
                          ((float)q.z - z) * s, ((float)q.w - z) * s };
                *(v4h*)(wdst + i * 4) = __builtin_convertvector(f, v4h);
            }
        }
        __syncthreads();

        // ---- 4 K-steps of v_wmma_f32_16x16x32_f16, 4 N-subtiles each ----
        #pragma unroll
        for (int ks = 0; ks < 4; ++ks) {
            const int k0 = ks * 32;
            v8h alo = *(const v8h*)&Ash[aRow * kKC + k0 + aKlo];
            v8h ahi = *(const v8h*)&Ash[aRow * kKC + k0 + aKlo + 8];
            v16h a  = __builtin_shufflevector(alo, ahi,
                        0,1,2,3,4,5,6,7,8,9,10,11,12,13,14,15);
            v16h b0 = *(const v16h*)&Wsh[(nHalf +  0 + ln) * kKC + k0 + bK];
            v16h b1 = *(const v16h*)&Wsh[(nHalf + 16 + ln) * kKC + k0 + bK];
            v16h b2 = *(const v16h*)&Wsh[(nHalf + 32 + ln) * kKC + k0 + bK];
            v16h b3 = *(const v16h*)&Wsh[(nHalf + 48 + ln) * kKC + k0 + bK];
            c0 = __builtin_amdgcn_wmma_f32_16x16x32_f16(false, a, false, b0, (short)0, c0, false, false);
            c1 = __builtin_amdgcn_wmma_f32_16x16x32_f16(false, a, false, b1, (short)0, c1, false, false);
            c2 = __builtin_amdgcn_wmma_f32_16x16x32_f16(false, a, false, b2, (short)0, c2, false, false);
            c3 = __builtin_amdgcn_wmma_f32_16x16x32_f16(false, a, false, b3, (short)0, c3, false, false);
        }
        __syncthreads();
    }

    // ============ LoRA-up epilogue (slice kz==0 only): K = 64 ============
    if (kz == 0) {
        // Reuse LDS: A2 = t[64x64] (stride 64), W2[o][l*16+r] = up[l][o][r].
        {
            const int row = tid >> 2;
            const int col = (tid & 3) * 16;
            const float* src = tws + row * kLR + col;
            _Float16*    dst = &Ash[row * kLR + col];
            #pragma unroll
            for (int i = 0; i < 4; ++i) {
                v4f f = *(const v4f*)(src + i * 4);
                *(v4h*)(dst + i * 4) = __builtin_convertvector(f, v4h);
            }
        }
        {
            const int wrow = tid >> 1;
            const int l0   = (tid & 1) * 2;       // adapters {0,1} or {2,3}
            const int o    = obase + wrow;
            _Float16* dst  = &Wsh[wrow * kLR + l0 * kR];
            #pragma unroll
            for (int l = 0; l < 2; ++l) {
                const float* src = up + ((size_t)(l0 + l) * kOUT + o) * kR;
                #pragma unroll
                for (int i = 0; i < 4; ++i) {
                    v4f f = *(const v4f*)(src + i * 4);
                    *(v4h*)(dst + l * kR + i * 4) = __builtin_convertvector(f, v4h);
                }
            }
        }
        __syncthreads();

        #pragma unroll
        for (int ks = 0; ks < 2; ++ks) {
            const int k0 = ks * 32;
            v8h alo = *(const v8h*)&Ash[aRow * kLR + k0 + aKlo];
            v8h ahi = *(const v8h*)&Ash[aRow * kLR + k0 + aKlo + 8];
            v16h a  = __builtin_shufflevector(alo, ahi,
                        0,1,2,3,4,5,6,7,8,9,10,11,12,13,14,15);
            v16h b0 = *(const v16h*)&Wsh[(nHalf +  0 + ln) * kLR + k0 + bK];
            v16h b1 = *(const v16h*)&Wsh[(nHalf + 16 + ln) * kLR + k0 + bK];
            v16h b2 = *(const v16h*)&Wsh[(nHalf + 32 + ln) * kLR + k0 + bK];
            v16h b3 = *(const v16h*)&Wsh[(nHalf + 48 + ln) * kLR + k0 + bK];
            c0 = __builtin_amdgcn_wmma_f32_16x16x32_f16(false, a, false, b0, (short)0, c0, false, false);
            c1 = __builtin_amdgcn_wmma_f32_16x16x32_f16(false, a, false, b1, (short)0, c1, false, false);
            c2 = __builtin_amdgcn_wmma_f32_16x16x32_f16(false, a, false, b2, (short)0, c2, false, false);
            c3 = __builtin_amdgcn_wmma_f32_16x16x32_f16(false, a, false, b3, (short)0, c3, false, false);
        }
    }

    // ---- store: C/D layout: VGPR j -> M = mBlk + lhi*8 + j, N = ln ----
    float* dst = out + (size_t)kz * kT * kOUT;
    #pragma unroll
    for (int j = 0; j < 8; ++j) {
        const int m = mBlk + lhi * 8 + j;
        float* yr = dst + (size_t)m * kOUT + obase + nHalf + ln;
        yr[ 0] = c0[j];
        yr[16] = c1[j];
        yr[32] = c2[j];
        yr[48] = c3[j];
    }
}

// ---------------------------------------------------------------------------
// Kernel 3: y = sum of KSPLIT partial slices (vectorized, 768 blocks)
// ---------------------------------------------------------------------------
__global__ __launch_bounds__(256)
void ksplit_reduce_kernel(const float* __restrict__ pws,
                          float* __restrict__ y)
{
    const size_t i = ((size_t)blockIdx.x * 256 + threadIdx.x) * 4;
    constexpr size_t slice = (size_t)kT * kOUT;
    v4f s = *(const v4f*)(pws + i);
    #pragma unroll
    for (int k = 1; k < kKSPLIT; ++k) {
        v4f p = *(const v4f*)(pws + k * slice + i);
        s.x += p.x; s.y += p.y; s.z += p.z; s.w += p.w;
    }
    *(v4f*)(y + i) = s;
}

// ---------------------------------------------------------------------------
extern "C" void kernel_launch(void* const* d_in, const int* in_sizes, int n_in,
                              void* d_out, int out_size, void* d_ws, size_t ws_size,
                              hipStream_t stream)
{
    const float* x      = (const float*)d_in[0];
    const int*   qw     = (const int*)  d_in[1];
    const int*   zeros  = (const int*)  d_in[2];
    const float* scales = (const float*)d_in[3];
    const float* up     = (const float*)d_in[4];
    const float* down   = (const float*)d_in[5];
    const float* ls     = (const float*)d_in[6];
    float* y   = (float*)d_out;
    float* tws = (float*)d_ws;                    // 64*64 f32 = 16 KB

    const size_t tBytes = (size_t)kT * kLR * sizeof(float);
    const size_t pBytes = (size_t)kKSPLIT * kT * kOUT * sizeof(float);

    lora_down_kernel<<<kLR, 256, 0, stream>>>(x, down, ls, tws);

    if (ws_size >= tBytes + pBytes) {
        // K-split path: 384 streaming blocks, deterministic two-phase reduce.
        float* pws = (float*)((char*)d_ws + tBytes);
        dim3 grid(kOUT / kONT, kKSPLIT);
        awq_lora_gemm<<<grid, 256, 0, stream>>>(x, qw, zeros, scales, up, tws,
                                                pws, (kIN / kKC) / kKSPLIT);
        const int nred = (kT * kOUT) / (256 * 4);
        ksplit_reduce_kernel<<<nred, 256, 0, stream>>>(pws, y);
    } else {
        // Fallback: single-slice direct write (identical math).
        dim3 grid(kOUT / kONT, 1);
        awq_lora_gemm<<<grid, 256, 0, stream>>>(x, qw, zeros, scales, up, tws,
                                                y, kIN / kKC);
    }
}
